// RoutingGNN_15625091023322
// MI455X (gfx1250) — compile-verified, matrix operand order checked by reading, not run on hardware
//
#include <hip/hip_runtime.h>
#include <hip/hip_bf16.h>

// RoutingGNN for MI455X (gfx1250, wave32, WMMA).
// Pipeline: 4x GAT layers (WMMA GEMM + atomic segment-softmax aggregation),
// fused 3-layer path MLP per 16-edge tile (WMMA in LDS), WMMA via/layer MLPs.

#define N_NODES   50000
#define N_EDGES   800000
#define EP_EDGES  (N_EDGES + N_NODES)   // edges + self loops
#define HID       128
#define NEG_SLOPE 0.2f

typedef __attribute__((ext_vector_type(16))) _Float16 v16h;
typedef __attribute__((ext_vector_type(8)))  float    v8f;

// ---- fragment swizzle (CDNA5 ISA §7.12.2, 16-bit A 16x32) ----
// K(v,j,half) = 16*(v>=4) + 8*half + 2*(v%4) + j ; lane = 16*half + r
// Fragment storage: [ktile][lane][16 halves], so a lane reads 32 contiguous B.
__device__ __forceinline__ void swz_store(_Float16* buf, int r, int k, float val) {
    int ktile = k >> 5, kk = k & 31;
    int j    = kk & 1;
    int vlow = (kk >> 1) & 3;
    int half = (kk >> 3) & 1;
    int v    = ((kk >> 4) & 1) * 4 + vlow;
    int lane = half * 16 + r;
    buf[(((ktile << 5) + lane) << 4) + (v << 1) + j] = (_Float16)val;
}

__device__ __forceinline__ void atomicMaxF(float* addr, float val) {
    // IEEE trick: non-negative -> int max; negative -> unsigned min
    if (val >= 0.f) atomicMax((int*)addr, __float_as_int(val));
    else            atomicMin((unsigned int*)addr, __float_as_uint(val));
}

__device__ __forceinline__ void edge_sd(const long long* ei, int e, int& s, int& d) {
    if (e < N_EDGES) { s = (int)ei[e]; d = (int)ei[N_EDGES + e]; }
    else             { s = e - N_EDGES; d = s; }              // self loop
}

// ---- one-time weight pre-swizzle: W[K][Nc] f32 -> [ntile][KT][lane][16] f16 --
__global__ void k_swizzle(const float* __restrict__ W, _Float16* __restrict__ out,
                          int K, int Nc) {
    int idx = blockIdx.x * blockDim.x + threadIdx.x;
    if (idx >= K * Nc) return;
    int k = idx / Nc, n = idx - k * Nc;
    int KT = K >> 5;
    int ntile = n >> 4, r = n & 15;
    int ktile = k >> 5, kk = k & 31;
    int j    = kk & 1;
    int vlow = (kk >> 1) & 3;
    int half = (kk >> 3) & 1;
    int v    = ((kk >> 4) & 1) * 4 + vlow;
    int lane = half * 16 + r;
    out[((((ntile * KT + ktile) << 5) + lane) << 4) + (v << 1) + j] = (_Float16)W[idx];
}

// ---- generic WMMA GEMM: C[M x NT*16] = act(A[M x KT*32] @ W + b) -----------
// grid = M/16 blocks, blockDim = NT*32 (one wave per 16x16 output tile)
template<int KT, int NT>
__global__ void k_wmma_gemm(const float* __restrict__ A,
                            const _Float16* __restrict__ Wswz,
                            const float* __restrict__ bias,
                            float* __restrict__ C, int relu) {
    constexpr int K = KT * 32;
    __shared__ __align__(32) _Float16 ldsA[KT * 32 * 16];
    const int tid = threadIdx.x;
    const int row0 = blockIdx.x << 4;
    for (int idx = tid; idx < 16 * K; idx += NT * 32) {
        int r = idx / K, k = idx - r * K;
        swz_store(ldsA, r, k, A[(row0 + r) * K + k]);
    }
    __syncthreads();
    const int w = tid >> 5, lane = tid & 31;
    v8f acc = {};
#pragma unroll
    for (int t = 0; t < KT; ++t) {
        v16h a = *(const v16h*)(ldsA + (((t << 5) + lane) << 4));
        v16h b = *(const v16h*)(Wswz + ((((w * KT + t) << 5) + lane) << 4));
        acc = __builtin_amdgcn_wmma_f32_16x16x32_f16(false, a, false, b,
                                                     (short)0, acc, false, false);
    }
    const int ldc = NT << 4;
    const int col = (w << 4) + (lane & 15);
    const int rowoff = (lane >> 4) << 3;
    const float bb = bias ? bias[col] : 0.f;
#pragma unroll
    for (int i = 0; i < 8; ++i) {
        float v = acc[i] + bb;
        if (relu) v = v > 0.f ? v : 0.f;
        C[(row0 + rowoff + i) * ldc + col] = v;
    }
}

// ---- GAT layer 0 GEMM (K=4, trivial) ---------------------------------------
__global__ void k_gemm0(const float* __restrict__ x, const float* __restrict__ W0,
                        float* __restrict__ z) {
    int idx = blockIdx.x * blockDim.x + threadIdx.x;
    if (idx >= N_NODES * HID) return;
    int n = idx >> 7, c = idx & 127;
    float4 xv = *(const float4*)(x + n * 4);
    z[idx] = xv.x * W0[c] + xv.y * W0[HID + c] + xv.z * W0[2 * HID + c]
           + xv.w * W0[3 * HID + c];
}

// ---- per-node attention logits: as[n]=z[n].a_s, ad[n]=z[n].a_d -------------
__global__ void k_alpha(const float* __restrict__ z, const float* __restrict__ a_s,
                        const float* __restrict__ a_d, float* __restrict__ as_,
                        float* __restrict__ ad_) {
    int n = blockIdx.x * blockDim.x + threadIdx.x;
    if (n >= N_NODES) return;
    const float4* zr = (const float4*)(z + n * HID);
    const float4* s4 = (const float4*)a_s;
    const float4* d4 = (const float4*)a_d;
    float ss = 0.f, dd = 0.f;
#pragma unroll 4
    for (int i = 0; i < HID / 4; ++i) {
        float4 zv = zr[i], sv = s4[i], dv = d4[i];
        ss += zv.x * sv.x + zv.y * sv.y + zv.z * sv.z + zv.w * sv.w;
        dd += zv.x * dv.x + zv.y * dv.y + zv.z * dv.z + zv.w * dv.w;
    }
    as_[n] = ss; ad_[n] = dd;
}

__global__ void k_init(float* __restrict__ hOut, float* __restrict__ m,
                       float* __restrict__ den) {
    int idx = blockIdx.x * blockDim.x + threadIdx.x;
    if (idx < N_NODES * HID) hOut[idx] = 0.f;
    if (idx < N_NODES) { m[idx] = -__builtin_inff(); den[idx] = 0.f; }
}

__global__ void k_edge_max(const long long* __restrict__ ei,
                           const float* __restrict__ as_,
                           const float* __restrict__ ad_, float* __restrict__ m) {
    int e = blockIdx.x * blockDim.x + threadIdx.x;
    if (e >= EP_EDGES) return;
    int s, d; edge_sd(ei, e, s, d);
    float v = as_[s] + ad_[d];
    v = v > 0.f ? v : NEG_SLOPE * v;
    atomicMaxF(&m[d], v);
}

__global__ void k_edge_exp(const long long* __restrict__ ei,
                           const float* __restrict__ as_,
                           const float* __restrict__ ad_,
                           const float* __restrict__ m, float* __restrict__ den,
                           float* __restrict__ eexp) {
    int e = blockIdx.x * blockDim.x + threadIdx.x;
    if (e >= EP_EDGES) return;
    int s, d; edge_sd(ei, e, s, d);
    float v = as_[s] + ad_[d];
    v = v > 0.f ? v : NEG_SLOPE * v;
    float ex = __expf(v - m[d]);
    eexp[e] = ex;
    atomicAdd(&den[d], ex);
}

// one wave32 per edge, lane covers 4 consecutive features
__global__ void k_edge_agg(const long long* __restrict__ ei,
                           const float* __restrict__ eexp,
                           const float* __restrict__ den,
                           const float* __restrict__ z, float* __restrict__ hOut) {
    int gid = blockIdx.x * blockDim.x + threadIdx.x;
    int e = gid >> 5, lane = gid & 31;
    if (e >= EP_EDGES) return;
    int s, d; edge_sd(ei, e, s, d);
    float alpha = eexp[e] / den[d];
    float4 zv = *(const float4*)(z + s * HID + (lane << 2));
    float* out = hOut + d * HID + (lane << 2);
    atomicAdd(out + 0, alpha * zv.x);
    atomicAdd(out + 1, alpha * zv.y);
    atomicAdd(out + 2, alpha * zv.z);
    atomicAdd(out + 3, alpha * zv.w);
}

__global__ void k_finalize(const float* __restrict__ hIn, const float* __restrict__ b,
                           float* __restrict__ dst, int relu) {
    int idx = blockIdx.x * blockDim.x + threadIdx.x;
    if (idx >= N_NODES * HID) return;
    float v = hIn[idx] + b[idx & 127];
    if (relu) v = v > 0.f ? v : 0.f;
    dst[idx] = v;
}

// ---- fused path MLP: [nf[src],nf[dst]](256) ->128 relu ->64 relu ->1 sigm --
// 16 edges per block, 8 waves; no global intermediates.
__global__ void k_path(const float* __restrict__ nf, const long long* __restrict__ ei,
                       const _Float16* __restrict__ W1s, const float* __restrict__ b1,
                       const _Float16* __restrict__ W2s, const float* __restrict__ b2,
                       const float* __restrict__ W3, const float* __restrict__ b3,
                       float* __restrict__ paths) {
    __shared__ __align__(32) _Float16 ldsA[8 * 32 * 16];   // 16 x 256 swizzled
    __shared__ __align__(32) _Float16 ldsB[4 * 32 * 16];   // 16 x 128 swizzled
    __shared__ float ldsC[16 * 64];
    const int tid = threadIdx.x;
    const int e0 = blockIdx.x << 4;
    for (int idx = tid; idx < 16 * 256; idx += 256) {
        int r = idx >> 8, k = idx & 255;
        int e = e0 + r;
        int node = (k < HID) ? (int)ei[e] : (int)ei[N_EDGES + e];
        swz_store(ldsA, r, k, nf[node * HID + (k & 127)]);
    }
    __syncthreads();
    const int w = tid >> 5, lane = tid & 31;
    const int col = (w << 4) + (lane & 15);
    const int rowoff = (lane >> 4) << 3;
    {   // stage 1: 256 -> 128, relu, write swizzled into ldsB
        v8f acc = {};
#pragma unroll
        for (int t = 0; t < 8; ++t) {
            v16h a = *(const v16h*)(ldsA + (((t << 5) + lane) << 4));
            v16h b = *(const v16h*)(W1s + ((((w << 3) + t) << 5) + lane) * 16);
            acc = __builtin_amdgcn_wmma_f32_16x16x32_f16(false, a, false, b,
                                                         (short)0, acc, false, false);
        }
        float bb = b1[col];
#pragma unroll
        for (int i = 0; i < 8; ++i) {
            float h = acc[i] + bb; h = h > 0.f ? h : 0.f;
            swz_store(ldsB, rowoff + i, col, h);
        }
    }
    __syncthreads();
    if (w < 4) {  // stage 2: 128 -> 64, relu (wave-uniform branch, EXEC full)
        v8f acc = {};
#pragma unroll
        for (int t = 0; t < 4; ++t) {
            v16h a = *(const v16h*)(ldsB + (((t << 5) + lane) << 4));
            v16h b = *(const v16h*)(W2s + ((((w << 2) + t) << 5) + lane) * 16);
            acc = __builtin_amdgcn_wmma_f32_16x16x32_f16(false, a, false, b,
                                                         (short)0, acc, false, false);
        }
        float bb = b2[col];
#pragma unroll
        for (int i = 0; i < 8; ++i) {
            float h = acc[i] + bb; h = h > 0.f ? h : 0.f;
            ldsC[(rowoff + i) * 64 + col] = h;
        }
    }
    __syncthreads();
    if (tid < 16) {  // stage 3: 64 -> 1, sigmoid
        float s = b3[0];
#pragma unroll 8
        for (int k = 0; k < 64; ++k) s += ldsC[tid * 64 + k] * W3[k];
        paths[e0 + tid] = 1.f / (1.f + __expf(-s));
    }
}

__global__ void k_via_final(const float* __restrict__ h, const float* __restrict__ W,
                            const float* __restrict__ b, float* __restrict__ vias) {
    int n = blockIdx.x * blockDim.x + threadIdx.x;
    if (n >= N_NODES) return;
    float s = b[0];
#pragma unroll 8
    for (int k = 0; k < 64; ++k) s += h[n * 64 + k] * W[k];
    vias[n] = 1.f / (1.f + __expf(-s));
}

__global__ void k_layer_final(const float* __restrict__ h, const float* __restrict__ W,
                              const float* __restrict__ b, float* __restrict__ outl) {
    int n = blockIdx.x * blockDim.x + threadIdx.x;
    if (n >= N_NODES) return;
    float v0 = b[0], v1 = b[1], v2 = b[2], v3 = b[3];
#pragma unroll 8
    for (int k = 0; k < 64; ++k) {
        float hv = h[n * 64 + k];
        v0 += hv * W[k * 4 + 0]; v1 += hv * W[k * 4 + 1];
        v2 += hv * W[k * 4 + 2]; v3 += hv * W[k * 4 + 3];
    }
    float mx = fmaxf(fmaxf(v0, v1), fmaxf(v2, v3));
    float e0 = __expf(v0 - mx), e1 = __expf(v1 - mx);
    float e2 = __expf(v2 - mx), e3 = __expf(v3 - mx);
    float inv = 1.f / (e0 + e1 + e2 + e3);
    outl[n * 4 + 0] = e0 * inv; outl[n * 4 + 1] = e1 * inv;
    outl[n * 4 + 2] = e2 * inv; outl[n * 4 + 3] = e3 * inv;
}

extern "C" void kernel_launch(void* const* d_in, const int* in_sizes, int n_in,
                              void* d_out, int out_size, void* d_ws, size_t ws_size,
                              hipStream_t stream) {
    (void)in_sizes; (void)n_in; (void)out_size; (void)ws_size;
    const float*     x  = (const float*)d_in[0];
    const long long* ei = (const long long*)d_in[1];   // int64 edge_index [2,E]
    const float *gatW[4], *gatAs[4], *gatAd[4], *gatB[4];
    for (int i = 0; i < 4; ++i) {
        gatW[i]  = (const float*)d_in[2 + 4 * i];
        gatAs[i] = (const float*)d_in[3 + 4 * i];
        gatAd[i] = (const float*)d_in[4 + 4 * i];
        gatB[i]  = (const float*)d_in[5 + 4 * i];
    }
    const float* pW1 = (const float*)d_in[18]; const float* pb1 = (const float*)d_in[19];
    const float* pW2 = (const float*)d_in[20]; const float* pb2 = (const float*)d_in[21];
    const float* pW3 = (const float*)d_in[22]; const float* pb3 = (const float*)d_in[23];
    const float* vW1 = (const float*)d_in[24]; const float* vb1 = (const float*)d_in[25];
    const float* vW2 = (const float*)d_in[26]; const float* vb2 = (const float*)d_in[27];
    const float* lW1 = (const float*)d_in[28]; const float* lb1 = (const float*)d_in[29];
    const float* lW2 = (const float*)d_in[30]; const float* lb2 = (const float*)d_in[31];

    // outputs (concatenated in return order)
    float* out_paths  = (float*)d_out;
    float* out_vias   = out_paths + N_EDGES;
    float* out_layers = out_vias + N_NODES;
    float* nf         = out_layers + (size_t)N_NODES * 4;  // node_features [N,128]

    // workspace (~107 MB)
    float* z     = (float*)d_ws;
    float* bufA  = z    + (size_t)N_NODES * HID;
    float* bufB  = bufA + (size_t)N_NODES * HID;
    float* as_   = bufB + (size_t)N_NODES * HID;
    float* ad_   = as_ + N_NODES;
    float* m_    = ad_ + N_NODES;
    float* den   = m_  + N_NODES;
    float* eexp  = den + N_NODES;
    float* via_h = eexp + EP_EDGES;
    float* lay_h = via_h + (size_t)N_NODES * 64;
    _Float16* sw_gat1 = (_Float16*)(lay_h + (size_t)N_NODES * 64);
    _Float16* sw_gat2 = sw_gat1 + 16384;
    _Float16* sw_gat3 = sw_gat2 + 16384;
    _Float16* sw_p1   = sw_gat3 + 16384;
    _Float16* sw_p2   = sw_p1 + 32768;
    _Float16* sw_v1   = sw_p2 + 8192;
    _Float16* sw_l1   = sw_v1 + 8192;

    auto SWZ = [&](const float* W, _Float16* o, int K, int Nc) {
        int tot = K * Nc;
        k_swizzle<<<dim3((tot + 255) / 256), dim3(256), 0, stream>>>(W, o, K, Nc);
    };
    SWZ(gatW[1], sw_gat1, 128, 128);
    SWZ(gatW[2], sw_gat2, 128, 128);
    SWZ(gatW[3], sw_gat3, 128, 128);
    SWZ(pW1, sw_p1, 256, 128);
    SWZ(pW2, sw_p2, 128, 64);
    SWZ(vW1, sw_v1, 128, 64);
    SWZ(lW1, sw_l1, 128, 64);

    const int NH = N_NODES * HID;
    dim3 b256(256);
    dim3 gNH((NH + 255) / 256);
    dim3 gN((N_NODES + 255) / 256);
    dim3 gE((EP_EDGES + 255) / 256);
    dim3 gAgg(((size_t)EP_EDGES * 32 + 255) / 256);

    const float*    gatIn[4]  = { x,    bufA, bufB, bufA };
    float*          gatOut[4] = { bufA, bufB, bufA, nf   };
    const _Float16* gatWs[4]  = { nullptr, sw_gat1, sw_gat2, sw_gat3 };

    for (int L = 0; L < 4; ++L) {
        if (L == 0)
            k_gemm0<<<gNH, b256, 0, stream>>>(x, gatW[0], z);
        else
            k_wmma_gemm<4, 8><<<dim3(N_NODES / 16), dim3(256), 0, stream>>>(
                gatIn[L], gatWs[L], nullptr, z, 0);
        k_alpha<<<gN, b256, 0, stream>>>(z, gatAs[L], gatAd[L], as_, ad_);
        float* hOut = gatOut[L];
        k_init<<<gNH, b256, 0, stream>>>(hOut, m_, den);
        k_edge_max<<<gE, b256, 0, stream>>>(ei, as_, ad_, m_);
        k_edge_exp<<<gE, b256, 0, stream>>>(ei, as_, ad_, m_, den, eexp);
        k_edge_agg<<<gAgg, b256, 0, stream>>>(ei, eexp, den, z, hOut);
        k_finalize<<<gNH, b256, 0, stream>>>(hOut, gatB[L], hOut, (L < 3) ? 1 : 0);
    }

    // heads
    k_path<<<dim3(N_EDGES / 16), dim3(256), 0, stream>>>(
        nf, ei, sw_p1, pb1, sw_p2, pb2, pW3, pb3, out_paths);
    k_wmma_gemm<4, 4><<<dim3(N_NODES / 16), dim3(128), 0, stream>>>(
        nf, sw_v1, vb1, via_h, 1);
    k_via_final<<<gN, b256, 0, stream>>>(via_h, vW2, vb2, out_vias);
    k_wmma_gemm<4, 4><<<dim3(N_NODES / 16), dim3(128), 0, stream>>>(
        nf, sw_l1, lb1, lay_h, 1);
    k_layer_final<<<gN, b256, 0, stream>>>(lay_h, lW2, lb2, out_layers);
}
